// Encoder_78176994721982
// MI455X (gfx1250) — matrix operand order repaired and checked by reading, not compiled, not run on hardware
//
#include <hip/hip_runtime.h>
#include <stdint.h>

typedef __attribute__((ext_vector_type(16))) __bf16 bf16x16;
typedef __attribute__((ext_vector_type(8)))  float  f32x8;

struct Frag { unsigned w[8]; };

static __device__ inline unsigned short bf16r(float x) {
  unsigned u = __float_as_uint(x);
  u += 0x7FFFu + ((u >> 16) & 1u);   // round-to-nearest-even
  return (unsigned short)(u >> 16);
}
static __device__ inline unsigned bf16pk(float a, float b) {
  return (unsigned)bf16r(a) | ((unsigned)bf16r(b) << 16);
}
static __device__ inline f32x8 wmma_bf16(const Frag& a, const Frag& b, f32x8 c) {
  return __builtin_amdgcn_wmma_f32_16x16x32_bf16(
      false, __builtin_bit_cast(bf16x16, a),
      false, __builtin_bit_cast(bf16x16, b),
      (short)0, c, false, false);
}
// 8 packed-bf16 dwords, 32B aligned (two b128 loads)
static __device__ inline Frag ldfrag(const unsigned* __restrict__ p) {
  Frag f;
  const uint4* q = (const uint4*)p;
  uint4 x = q[0], y = q[1];
  f.w[0] = x.x; f.w[1] = x.y; f.w[2] = x.z; f.w[3] = x.w;
  f.w[4] = y.x; f.w[5] = y.y; f.w[6] = y.z; f.w[7] = y.w;
  return f;
}
// 8 contiguous halfs -> 4 dwords (one b128 load)
static __device__ inline void ld_half8(const unsigned short* p, unsigned* dst) {
  uint4 x = *(const uint4*)p;
  dst[0] = x.x; dst[1] = x.y; dst[2] = x.z; dst[3] = x.w;
}
static __device__ inline float eluf(float v) { return v > 0.f ? v : (__expf(v) - 1.f); }

// ---------------- weight pre-pack into WMMA B-fragment order -----------------
// dst[((kc*NB+nb)*32 + lane)*8 + v] packs W[K][n], W[K+1][n] as bf16 pair,
// K = kc*32 + (lane>>4)*8 + (v>>2)*8 + 2v, n = nb*16 + (lane&15); zero-padded past Korig.
__global__ void pack_w_kernel(const float* __restrict__ W, unsigned* __restrict__ dst,
                              int Korig, int Ncols, int KC, int NB) {
  int idx = blockIdx.x * 256 + threadIdx.x;
  int total = KC * NB * 256;
  if (idx >= total) return;
  int frag = idx >> 8;
  int rem = idx & 255;
  int lane = rem >> 3;
  int v = rem & 7;
  int kc = frag / NB, nb = frag % NB;
  int K = kc * 32 + (lane >> 4) * 8 + (v >> 2) * 8 + 2 * v;
  int n = nb * 16 + (lane & 15);
  float lo = (K     < Korig) ? W[(size_t)K * Ncols + n]       : 0.f;
  float hi = (K + 1 < Korig) ? W[(size_t)(K + 1) * Ncols + n] : 0.f;
  dst[idx] = bf16pk(lo, hi);
}

// ---------------- h = h0 @ emb_w + emb_b  -> bf16 ---------------------------
__global__ void embed_kernel(const float* __restrict__ h0, const float* __restrict__ emb_w,
                             const float* __restrict__ emb_b, unsigned short* __restrict__ hbf,
                             int N) {
  long idx = (long)blockIdx.x * 256 + threadIdx.x;
  if (idx >= (long)N * 128) return;
  int n = (int)(idx >> 7), c = (int)(idx & 127);
  float s = emb_b[c];
#pragma unroll
  for (int k = 0; k < 11; k++) s += h0[(size_t)n * 11 + k] * emb_w[k * 128 + c];
  hbf[idx] = bf16r(s);
}

__global__ void radial_kernel(const float* __restrict__ x, const int* __restrict__ rowi,
                              const int* __restrict__ coli, float* __restrict__ radial, int E) {
  int e = blockIdx.x * 256 + threadIdx.x;
  if (e >= E) return;
  int r = rowi[e], c = coli[e];
  float dx = x[(size_t)r * 3 + 0] - x[(size_t)c * 3 + 0];
  float dy = x[(size_t)r * 3 + 1] - x[(size_t)c * 3 + 1];
  float dz = x[(size_t)r * 3 + 2] - x[(size_t)c * 3 + 2];
  radial[e] = dx * dx + dy * dy + dz * dz;
}

__global__ void labpack_kernel(const float* __restrict__ label, unsigned short* __restrict__ labbf,
                               int N) {
  long idx = (long)blockIdx.x * 256 + threadIdx.x;
  if (idx >= (long)N * 8) return;
  int n = (int)(idx >> 3), k = (int)(idx & 7);
  labbf[idx] = bf16r(k < 7 ? label[(size_t)n * 7 + k] : 0.f);
}

// ---------------- edge model: fused 2-layer MLP + atomic scatter -------------
// One wave32 per PAIR of 16-edge tiles (32 edges): each B fragment is loaded
// once and reused for two WMMAs, halving L2 weight traffic.
// K1 = 288 (padded 261), K2 = 128.
__global__ __launch_bounds__(128) void edge_kernel(
    const unsigned short* __restrict__ hbf, const float* __restrict__ radial,
    const float* __restrict__ eattr, const int* __restrict__ rowi, const int* __restrict__ coli,
    const unsigned* __restrict__ w1f, const unsigned* __restrict__ w2f,
    const float* __restrict__ b1, const float* __restrict__ b2,
    float* __restrict__ agg, int E) {
  __shared__ unsigned short lds[4][2][16 * 128];
  const int lane = threadIdx.x & 31;
  const int wid = threadIdx.x >> 5;
  const int pair = blockIdx.x * 4 + wid;
  const int eBase = pair * 32;
  const int mrow = lane & 15;
  const int hi = lane >> 4;

  int e0 = eBase + mrow;
  int e1 = eBase + 16 + mrow;
  int e0c = (e0 < E) ? e0 : (E - 1);
  int e1c = (e1 < E) ? e1 : (E - 1);
  int rA0 = rowi[e0c], cA0 = coli[e0c];
  int rA1 = rowi[e1c], cA1 = coli[e1c];

  const f32x8 z8 = {0.f, 0.f, 0.f, 0.f, 0.f, 0.f, 0.f, 0.f};
  f32x8 acc0[8], acc1[8];
#pragma unroll
  for (int nb = 0; nb < 8; nb++) { acc0[nb] = z8; acc1[nb] = z8; }

  for (int kc = 0; kc < 9; kc++) {
    Frag a0, a1;
    if (kc < 8) {
      int nd0 = (kc < 4) ? rA0 : cA0;
      int nd1 = (kc < 4) ? rA1 : cA1;
      int ko = (kc & 3) * 32 + hi * 8;
      const unsigned short* p0 = hbf + (size_t)nd0 * 128 + ko;
      ld_half8(p0, &a0.w[0]);
      ld_half8(p0 + 16, &a0.w[4]);
      const unsigned short* p1 = hbf + (size_t)nd1 * 128 + ko;
      ld_half8(p1, &a1.w[0]);
      ld_half8(p1 + 16, &a1.w[4]);
    } else {  // K 256..287: [radial, ea0..ea3, zeros...]
#pragma unroll
      for (int i = 0; i < 8; i++) { a0.w[i] = 0u; a1.w[i] = 0u; }
      if (hi == 0) {
        float r0 = radial[e0c];
        float4 ea0 = *(const float4*)(eattr + (size_t)e0c * 4);
        a0.w[0] = bf16pk(r0, ea0.x);
        a0.w[1] = bf16pk(ea0.y, ea0.z);
        a0.w[2] = bf16pk(ea0.w, 0.f);
        float r1 = radial[e1c];
        float4 ea1 = *(const float4*)(eattr + (size_t)e1c * 4);
        a1.w[0] = bf16pk(r1, ea1.x);
        a1.w[1] = bf16pk(ea1.y, ea1.z);
        a1.w[2] = bf16pk(ea1.w, 0.f);
      }
    }
    const unsigned* wp = w1f + (size_t)(kc * 8) * 256;
#pragma unroll
    for (int nb = 0; nb < 8; nb++) {
      Frag b = ldfrag(wp + (size_t)nb * 256 + lane * 8);
      acc0[nb] = wmma_bf16(a0, b, acc0[nb]);
      acc1[nb] = wmma_bf16(a1, b, acc1[nb]);
    }
  }
  // bias + ELU, C-layout -> row-major bf16 LDS tiles (one per M-tile)
  {
    unsigned short* l0 = &lds[wid][0][0];
    unsigned short* l1 = &lds[wid][1][0];
#pragma unroll
    for (int nb = 0; nb < 8; nb++) {
      float bias = b1[nb * 16 + mrow];
#pragma unroll
      for (int i = 0; i < 8; i++) {
        float v0 = eluf(acc0[nb][i] + bias);
        l0[(i + 8 * hi) * 128 + nb * 16 + mrow] = bf16r(v0);
        float v1 = eluf(acc1[nb][i] + bias);
        l1[(i + 8 * hi) * 128 + nb * 16 + mrow] = bf16r(v1);
      }
    }
  }
  __syncthreads();

  f32x8 acc2a[8], acc2b[8];
#pragma unroll
  for (int nb = 0; nb < 8; nb++) { acc2a[nb] = z8; acc2b[nb] = z8; }
  for (int kc = 0; kc < 4; kc++) {
    Frag a0, a1;
    const unsigned short* p0 = &lds[wid][0][0] + mrow * 128 + kc * 32 + hi * 8;
    ld_half8(p0, &a0.w[0]);
    ld_half8(p0 + 16, &a0.w[4]);
    const unsigned short* p1 = &lds[wid][1][0] + mrow * 128 + kc * 32 + hi * 8;
    ld_half8(p1, &a1.w[0]);
    ld_half8(p1 + 16, &a1.w[4]);
    const unsigned* wp = w2f + (size_t)(kc * 8) * 256;
#pragma unroll
    for (int nb = 0; nb < 8; nb++) {
      Frag b = ldfrag(wp + (size_t)nb * 256 + lane * 8);
      acc2a[nb] = wmma_bf16(a0, b, acc2a[nb]);
      acc2b[nb] = wmma_bf16(a1, b, acc2b[nb]);
    }
  }
  // bias + ELU + segment_sum scatter (fp32 atomics), both M-tiles
#pragma unroll
  for (int t = 0; t < 2; t++) {
    const f32x8* a2 = (t == 0) ? acc2a : acc2b;
#pragma unroll
    for (int i = 0; i < 8; i++) {
      int eo = eBase + t * 16 + i + 8 * hi;
      if (eo >= E) continue;
      int node = rowi[eo];
      float* ap = agg + (size_t)node * 128;
#pragma unroll
      for (int nb = 0; nb < 8; nb++) {
        float v = eluf(a2[nb][i] + b2[nb * 16 + mrow]);
        __hip_atomic_fetch_add(ap + nb * 16 + mrow, v, __ATOMIC_RELAXED,
                               __HIP_MEMORY_SCOPE_AGENT);
      }
    }
  }
}

// ---------------- node model: nf=[h,agg,h] -> elu(GEMM) -> GEMM, update hbf --
__global__ __launch_bounds__(32) void node_kernel(
    unsigned short* __restrict__ hbf, const float* __restrict__ agg,
    const unsigned* __restrict__ w1f, const unsigned* __restrict__ w2f,
    const float* __restrict__ b1, const float* __restrict__ b2, int N) {
  __shared__ unsigned short lds[16 * 128];
  const int lane = threadIdx.x & 31;
  const int mrow = lane & 15;
  const int hi = lane >> 4;
  const int nBase = blockIdx.x * 16;
  int nd = nBase + mrow;
  if (nd >= N) nd = N - 1;

  const f32x8 z8 = {0.f, 0.f, 0.f, 0.f, 0.f, 0.f, 0.f, 0.f};
  f32x8 acc[8];
#pragma unroll
  for (int nb = 0; nb < 8; nb++) acc[nb] = z8;

  for (int kc = 0; kc < 12; kc++) {
    Frag a;
    if (kc < 4 || kc >= 8) {
      int ko = ((kc < 4) ? kc : (kc - 8)) * 32;
      const unsigned short* p = hbf + (size_t)nd * 128 + ko + hi * 8;
      ld_half8(p, &a.w[0]);
      ld_half8(p + 16, &a.w[4]);
    } else {  // agg fp32 -> bf16 on the fly
      const float* g = agg + (size_t)nd * 128 + (kc - 4) * 32 + hi * 8;
      float4 f0 = *(const float4*)g, f1 = *(const float4*)(g + 4);
      float4 f2 = *(const float4*)(g + 16), f3 = *(const float4*)(g + 20);
      a.w[0] = bf16pk(f0.x, f0.y); a.w[1] = bf16pk(f0.z, f0.w);
      a.w[2] = bf16pk(f1.x, f1.y); a.w[3] = bf16pk(f1.z, f1.w);
      a.w[4] = bf16pk(f2.x, f2.y); a.w[5] = bf16pk(f2.z, f2.w);
      a.w[6] = bf16pk(f3.x, f3.y); a.w[7] = bf16pk(f3.z, f3.w);
    }
    const unsigned* wp = w1f + (size_t)(kc * 8) * 256;
#pragma unroll
    for (int nb = 0; nb < 8; nb++) {
      Frag b = ldfrag(wp + (size_t)nb * 256 + lane * 8);
      acc[nb] = wmma_bf16(a, b, acc[nb]);
    }
  }
#pragma unroll
  for (int nb = 0; nb < 8; nb++) {
    float bias = b1[nb * 16 + mrow];
#pragma unroll
    for (int i = 0; i < 8; i++) {
      float v = eluf(acc[nb][i] + bias);
      lds[(i + 8 * hi) * 128 + nb * 16 + mrow] = bf16r(v);
    }
  }
  __syncthreads();

  f32x8 acc2[8];
#pragma unroll
  for (int nb = 0; nb < 8; nb++) acc2[nb] = z8;
  for (int kc = 0; kc < 4; kc++) {
    Frag a;
    const unsigned short* p = lds + mrow * 128 + kc * 32 + hi * 8;
    ld_half8(p, &a.w[0]);
    ld_half8(p + 16, &a.w[4]);
    const unsigned* wp = w2f + (size_t)(kc * 8) * 256;
#pragma unroll
    for (int nb = 0; nb < 8; nb++) {
      Frag b = ldfrag(wp + (size_t)nb * 256 + lane * 8);
      acc2[nb] = wmma_bf16(a, b, acc2[nb]);
    }
  }
  // h = nf@w2 + b2 (no activation), write back bf16 (in-place safe: per-node rows)
#pragma unroll
  for (int i = 0; i < 8; i++) {
    int no = nBase + i + 8 * hi;
    if (no >= N) continue;
#pragma unroll
    for (int nb = 0; nb < 8; nb++) {
      float v = acc2[nb][i] + b2[nb * 16 + mrow];
      hbf[(size_t)no * 128 + nb * 16 + mrow] = bf16r(v);
    }
  }
}

// ---------------- head: hl=[h,label] -> mu/var GEMMs + reparameterize --------
__global__ __launch_bounds__(32) void head_kernel(
    const unsigned short* __restrict__ hbf, const unsigned short* __restrict__ labbf,
    const float* __restrict__ eps, const unsigned* __restrict__ muf,
    const unsigned* __restrict__ varf, const float* __restrict__ mub,
    const float* __restrict__ varb, float* __restrict__ out, int N) {
  const int lane = threadIdx.x & 31;
  const int mrow = lane & 15;
  const int hi = lane >> 4;
  const int nBase = blockIdx.x * 16;
  int nd = nBase + mrow;
  if (nd >= N) nd = N - 1;

  const f32x8 z8 = {0.f, 0.f, 0.f, 0.f, 0.f, 0.f, 0.f, 0.f};
  f32x8 am[4], av[4];
#pragma unroll
  for (int nb = 0; nb < 4; nb++) { am[nb] = z8; av[nb] = z8; }

  for (int kc = 0; kc < 5; kc++) {
    Frag a;
    if (kc < 4) {
      const unsigned short* p = hbf + (size_t)nd * 128 + kc * 32 + hi * 8;
      ld_half8(p, &a.w[0]);
      ld_half8(p + 16, &a.w[4]);
    } else {  // K 128..159: label (7) + zero pad
#pragma unroll
      for (int i = 0; i < 8; i++) a.w[i] = 0u;
      if (hi == 0) {
        uint4 L = *(const uint4*)(labbf + (size_t)nd * 8);
        a.w[0] = L.x; a.w[1] = L.y; a.w[2] = L.z; a.w[3] = L.w;
      }
    }
#pragma unroll
    for (int nb = 0; nb < 4; nb++) {
      Frag bm = ldfrag(muf + (size_t)(kc * 4 + nb) * 256 + lane * 8);
      am[nb] = wmma_bf16(a, bm, am[nb]);
      Frag bv = ldfrag(varf + (size_t)(kc * 4 + nb) * 256 + lane * 8);
      av[nb] = wmma_bf16(a, bv, av[nb]);
    }
  }
#pragma unroll
  for (int i = 0; i < 8; i++) {
    int no = nBase + i + 8 * hi;
    if (no >= N) continue;
#pragma unroll
    for (int nb = 0; nb < 4; nb++) {
      int n = nb * 16 + mrow;
      float mu = am[nb][i] + mub[n];
      float lv = av[nb][i] + varb[n];
      float z = mu + 0.01f * eps[(size_t)no * 64 + n] * __expf(0.5f * lv);
      out[(size_t)no * 64 + n] = z;
    }
  }
}

// -----------------------------------------------------------------------------
extern "C" void kernel_launch(void* const* d_in, const int* in_sizes, int n_in,
                              void* d_out, int out_size, void* d_ws, size_t ws_size,
                              hipStream_t stream) {
  const float* h0      = (const float*)d_in[0];
  const float* label   = (const float*)d_in[1];
  const float* x       = (const float*)d_in[2];
  const float* eattr   = (const float*)d_in[3];
  const float* eps     = (const float*)d_in[4];
  const float* emb_w   = (const float*)d_in[5];
  const float* emb_b   = (const float*)d_in[6];
  const float* edge_w1 = (const float*)d_in[7];
  const float* edge_b1 = (const float*)d_in[8];
  const float* edge_w2 = (const float*)d_in[9];
  const float* edge_b2 = (const float*)d_in[10];
  const float* node_w1 = (const float*)d_in[11];
  const float* node_b1 = (const float*)d_in[12];
  const float* node_w2 = (const float*)d_in[13];
  const float* node_b2 = (const float*)d_in[14];
  const float* mu_w    = (const float*)d_in[15];
  const float* mu_b    = (const float*)d_in[16];
  const float* var_w   = (const float*)d_in[17];
  const float* var_b   = (const float*)d_in[18];
  const int*   edges   = (const int*)d_in[19];

  const int N = in_sizes[0] / 11;
  const int E = in_sizes[3] / 4;
  const int* rowi = edges;
  const int* coli = edges + E;
  float* out = (float*)d_out;
  (void)n_in; (void)out_size; (void)ws_size;

  char* base = (char*)d_ws;
  size_t off = 0;
  auto take = [&](size_t bytes) -> char* {
    char* p = base + off;
    off = (off + bytes + 255) & ~(size_t)255;
    return p;
  };
  unsigned short* hbf    = (unsigned short*)take((size_t)N * 128 * 2);
  unsigned short* labbf  = (unsigned short*)take((size_t)N * 8 * 2);
  float*          radial = (float*)take((size_t)E * 4);
  float*          agg    = (float*)take((size_t)N * 128 * 4);
  unsigned* ew1f[2]; unsigned* ew2f[2]; unsigned* nw1f[2]; unsigned* nw2f[2];
  for (int l = 0; l < 2; l++) {
    ew1f[l] = (unsigned*)take((size_t)9 * 8 * 256 * 4);
    ew2f[l] = (unsigned*)take((size_t)4 * 8 * 256 * 4);
    nw1f[l] = (unsigned*)take((size_t)12 * 8 * 256 * 4);
    nw2f[l] = (unsigned*)take((size_t)4 * 8 * 256 * 4);
  }
  unsigned* muf  = (unsigned*)take((size_t)5 * 4 * 256 * 4);
  unsigned* varf = (unsigned*)take((size_t)5 * 4 * 256 * 4);

  auto packW = [&](const float* W, unsigned* dst, int Korig, int Ncols, int KC, int NB) {
    int tot = KC * NB * 256;
    hipLaunchKernelGGL(pack_w_kernel, dim3((tot + 255) / 256), dim3(256), 0, stream,
                       W, dst, Korig, Ncols, KC, NB);
  };
  for (int l = 0; l < 2; l++) {
    packW(edge_w1 + (size_t)l * 261 * 128, ew1f[l], 261, 128, 9, 8);
    packW(edge_w2 + (size_t)l * 128 * 128, ew2f[l], 128, 128, 4, 8);
    packW(node_w1 + (size_t)l * 384 * 128, nw1f[l], 384, 128, 12, 8);
    packW(node_w2 + (size_t)l * 128 * 128, nw2f[l], 128, 128, 4, 8);
  }
  packW(mu_w, muf, 135, 64, 5, 4);
  packW(var_w, varf, 135, 64, 5, 4);

  {
    long tot = (long)N * 128;
    hipLaunchKernelGGL(embed_kernel, dim3((unsigned)((tot + 255) / 256)), dim3(256), 0, stream,
                       h0, emb_w, emb_b, hbf, N);
  }
  hipLaunchKernelGGL(radial_kernel, dim3((E + 255) / 256), dim3(256), 0, stream,
                     x, rowi, coli, radial, E);
  hipLaunchKernelGGL(labpack_kernel, dim3((unsigned)(((long)N * 8 + 255) / 256)), dim3(256), 0,
                     stream, label, labbf, N);

  const int tilesE = (E + 15) / 16;
  const int pairsE = (tilesE + 1) / 2;     // 32-edge pairs, one per wave
  const int tilesN = (N + 15) / 16;
  for (int l = 0; l < 2; l++) {
    hipMemsetAsync(agg, 0, (size_t)N * 128 * 4, stream);
    hipLaunchKernelGGL(edge_kernel, dim3((pairsE + 3) / 4), dim3(128), 0, stream,
                       hbf, radial, eattr, rowi, coli, ew1f[l], ew2f[l],
                       edge_b1 + l * 128, edge_b2 + l * 128, agg, E);
    hipLaunchKernelGGL(node_kernel, dim3(tilesN), dim3(32), 0, stream,
                       hbf, agg, nw1f[l], nw2f[l], node_b1 + l * 128, node_b2 + l * 128, N);
  }
  hipLaunchKernelGGL(head_kernel, dim3(tilesN), dim3(32), 0, stream,
                     hbf, labbf, eps, muf, varf, mu_b, var_b, out, N);
}
